// TaskDAGEncoder_16690242912871
// MI455X (gfx1250) — matrix operand order repaired and checked by reading, not compiled, not run on hardware
//
#include <hip/hip_runtime.h>
#include <hip/hip_bf16.h>

// ---------------------------------------------------------------------------
// Bidirectional GraphSAGE x2 + BN/ReLU + column max, for MI455X (gfx1250).
// - GEMMs: v_wmma_f32_16x16x32_bf16 (wave32, f32 accumulate), three products
//   per layer fused into one pass over the row data.
// - A tiles staged per-block into LDS as bf16 (ds_load_b128 fragments),
//   mean-normalization folded into staging.
// - B (weights) pre-packed once per launch into WMMA fragment order (bf16),
//   so B fragments are single aligned 32B loads from L2.
// - Edge aggregation: f32 global atomics (features are L2-resident, 192MB).
// ---------------------------------------------------------------------------

typedef __attribute__((ext_vector_type(16))) __bf16 v16bf;
typedef __attribute__((ext_vector_type(8)))  float  v8f;

#define HID 256
#define BN_EPS 1e-5f
#define ROWS_PER_BLK 128

__device__ __forceinline__ v8f wmma_bf16(v16bf a, v16bf b, v8f c) {
  return __builtin_amdgcn_wmma_f32_16x16x32_bf16(false, a, false, b,
                                                 (short)0, c, false, false);
}

// A fragment from LDS (bf16 row-major [16][DIN]); base = &row(lane&15).
// ISA 7.12.2 16-bit A 16x32: lanes 0-15 K = {e, 16+e}, lanes 16-31 +8.
__device__ __forceinline__ v16bf lds_a_frag(const __bf16* base, int kb, int half) {
  const __bf16* p = base + kb + half * 8;
  v16bf a;
#pragma unroll
  for (int e = 0; e < 8; ++e) {
    a[e]     = p[e];        // 16B aligned -> ds_load_b128
    a[8 + e] = p[16 + e];   // +32B        -> ds_load_b128
  }
  return a;
}

// ---------------------------------------------------------------------------
// Pack W (row-major [DIN,256] f32, optionally W0+W1) into bf16 WMMA B-fragment
// order: frag index = (nt*(DIN/32)+kbi)*32 + lane, 16 contiguous bf16 each.
// B 32x16 layout: lane = half*16 + (n%16); lanes 0-15 K = {e, 16+e}, lanes 16-31 +8.
// ---------------------------------------------------------------------------
template <int DIN>
__global__ void pack_b(const float* __restrict__ W0, const float* __restrict__ W1,
                       __bf16* __restrict__ out) {
  const int idx = blockIdx.x * blockDim.x + threadIdx.x;  // over DIN*HID
  if (idx >= DIN * HID) return;
  const int k = idx / HID;
  const int n = idx - k * HID;
  float v = W0[idx];
  if (W1) v += W1[idx];
  const int nt = n >> 4, nn = n & 15;
  const int kbi = k >> 5, krem = k & 31;
  const int half = (krem >> 3) & 1, e = (krem & 7) + ((krem >> 4) << 3);
  const int lane = half * 16 + nn;
  out[(((size_t)nt * (DIN / 32) + kbi) * 32 + lane) * 16 + e] = (__bf16)v;
}

// ---------------------------------------------------------------------------
// H = (aggF/cntF)@WlF + (aggB/cntB)@WlB + X@(WrF+WrB) + (blF+blB)
// Block = 256 threads = 8 waves; wave w owns N-tiles {2w, 2w+1} of the
// 16-row M-tile blockIdx.x. A tiles staged in LDS as bf16, shared by waves.
// ---------------------------------------------------------------------------
template <int DIN>
__global__ __launch_bounds__(256) void sage_gemm_wmma(
    const float* __restrict__ aggF, const float* __restrict__ aggB,
    const float* __restrict__ X,
    const float* __restrict__ cntF, const float* __restrict__ cntB,
    const __bf16* __restrict__ pWlF, const __bf16* __restrict__ pWlB,
    const __bf16* __restrict__ pWr,
    const float* __restrict__ blF, const float* __restrict__ blB,
    float* __restrict__ H) {
  constexpr int KSTEPS = DIN / 32;
  __shared__ __bf16 smA[3][16 * DIN];
  __shared__ float scl[2][16];

  const int tid   = threadIdx.x;
  const int wave  = tid >> 5;
  const int lane  = tid & 31;
  const int half  = lane >> 4;
  const int mtile = blockIdx.x;

  if (tid < 32) {  // per-row mean denominators
    const int r = tid & 15;
    const float c = ((tid < 16) ? cntF : cntB)[mtile * 16 + r];
    scl[tid >> 4][r] = 1.0f / fmaxf(c, 1.0f);
  }
  __syncthreads();

  // Stage three 16xDIN tiles to LDS as bf16 (coalesced b128 global loads).
#pragma unroll
  for (int m = 0; m < 3; ++m) {
    const float* P = (m == 0) ? aggF : (m == 1) ? aggB : X;
    for (int i = tid * 4; i < 16 * DIN; i += 256 * 4) {
      const int r = i / DIN;
      const int k = i - r * DIN;
      const float s = (m == 0) ? scl[0][r] : (m == 1) ? scl[1][r] : 1.0f;
      const float4 v = *(const float4*)(P + (size_t)(mtile * 16 + r) * DIN + k);
      __bf16* q = &smA[m][r * DIN + k];
      q[0] = (__bf16)(v.x * s);
      q[1] = (__bf16)(v.y * s);
      q[2] = (__bf16)(v.z * s);
      q[3] = (__bf16)(v.w * s);
    }
  }
  __syncthreads();

  const int nt0 = wave * 2;
  const int n0  = nt0 * 16 + (lane & 15);
  const int n1  = n0 + 16;

  const __bf16* aF = &smA[0][(lane & 15) * DIN];
  const __bf16* aB = &smA[1][(lane & 15) * DIN];
  const __bf16* aX = &smA[2][(lane & 15) * DIN];
  const v16bf* BF = (const v16bf*)pWlF;
  const v16bf* BB = (const v16bf*)pWlB;
  const v16bf* BR = (const v16bf*)pWr;

  v8f c0 = {};
  v8f c1 = {};

#pragma unroll
  for (int kbi = 0; kbi < KSTEPS; ++kbi) {
    const int kb = kbi * 32;
    const int i0 = (nt0 * KSTEPS + kbi) * 32 + lane;
    const int i1 = ((nt0 + 1) * KSTEPS + kbi) * 32 + lane;

    v16bf a = lds_a_frag(aF, kb, half);
    c0 = wmma_bf16(a, BF[i0], c0);
    c1 = wmma_bf16(a, BF[i1], c1);

    a = lds_a_frag(aB, kb, half);
    c0 = wmma_bf16(a, BB[i0], c0);
    c1 = wmma_bf16(a, BB[i1], c1);

    a = lds_a_frag(aX, kb, half);
    c0 = wmma_bf16(a, BR[i0], c0);
    c1 = wmma_bf16(a, BR[i1], c1);
  }

  const float bias0 = blF[n0] + blB[n0];
  const float bias1 = blF[n1] + blB[n1];
#pragma unroll
  for (int r = 0; r < 8; ++r) {
    const int mrow = mtile * 16 + r + 8 * half;  // C/D layout (ISA 7.12.2)
    H[(size_t)mrow * HID + n0] = c0[r] + bias0;
    H[(size_t)mrow * HID + n1] = c1[r] + bias1;
  }
}

// --------------------------- edge kernels ----------------------------------
__global__ void edge_count(const int* __restrict__ src, const int* __restrict__ dst,
                           float* cntF, float* cntB, int nE) {
  int e = blockIdx.x * blockDim.x + threadIdx.x;
  if (e < nE) {
    atomicAdd(&cntF[dst[e]], 1.0f);  // fwd: segment over dst
    atomicAdd(&cntB[src[e]], 1.0f);  // bwd: segment over src
  }
}

template <int DIN>
__global__ void edge_scatter(const int* __restrict__ src, const int* __restrict__ dst,
                             const float* __restrict__ X,
                             float* aggF, float* aggB, int nE) {
  const int e = blockIdx.x;
  const int k = threadIdx.x;  // DIN threads per block
  const int s = src[e];
  const int d = dst[e];
  atomicAdd(&aggF[(size_t)d * DIN + k], X[(size_t)s * DIN + k]);
  atomicAdd(&aggB[(size_t)s * DIN + k], X[(size_t)d * DIN + k]);
}

// --------------------------- batchnorm / max -------------------------------
__global__ void col_stats(const float* __restrict__ H, float* gsum,
                          float* gsumsq, int nRows) {
  const int col = threadIdx.x;  // 256 threads = 1 column each
  const int r0 = blockIdx.x * ROWS_PER_BLK;
  int r1 = r0 + ROWS_PER_BLK;
  if (r1 > nRows) r1 = nRows;
  float s = 0.0f, sq = 0.0f;
  for (int r = r0; r < r1; ++r) {
    const float v = H[(size_t)r * HID + col];
    s += v;
    sq += v * v;
  }
  atomicAdd(&gsum[col], s);
  atomicAdd(&gsumsq[col], sq);
}

__global__ void bn_finalize(const float* gsum, const float* gsumsq,
                            const float* __restrict__ gamma,
                            const float* __restrict__ beta,
                            float* scaleA, float* shiftB, int nRows) {
  const int c = threadIdx.x;
  const float inv = 1.0f / (float)nRows;
  const float mu = gsum[c] * inv;
  const float var = gsumsq[c] * inv - mu * mu;  // biased batch variance
  const float a = rsqrtf(var + BN_EPS) * gamma[c];
  scaleA[c] = a;
  shiftB[c] = beta[c] - mu * a;
}

__global__ void bn_apply_relu(float* __restrict__ H, const float* __restrict__ A,
                              const float* __restrict__ Bc, size_t total) {
  const size_t stride = (size_t)gridDim.x * blockDim.x;
  for (size_t i = (size_t)blockIdx.x * blockDim.x + threadIdx.x; i < total;
       i += stride) {
    const int c = (int)(i & (HID - 1));
    H[i] = fmaxf(fmaf(H[i], A[c], Bc[c]), 0.0f);
  }
}

__global__ void bn_relu_colmax(const float* __restrict__ H,
                               const float* __restrict__ A,
                               const float* __restrict__ Bc,
                               float* __restrict__ out, int nRows) {
  const int col = threadIdx.x;
  const float a = A[col], b = Bc[col];
  const int r0 = blockIdx.x * ROWS_PER_BLK;
  int r1 = r0 + ROWS_PER_BLK;
  if (r1 > nRows) r1 = nRows;
  float m = 0.0f;  // post-ReLU values are >= 0
  for (int r = r0; r < r1; ++r)
    m = fmaxf(m, fmaxf(fmaf(H[(size_t)r * HID + col], a, b), 0.0f));
  atomicMax((unsigned int*)(out + col), __float_as_uint(m));
}

// ---------------------------------------------------------------------------
extern "C" void kernel_launch(void* const* d_in, const int* in_sizes, int n_in,
                              void* d_out, int out_size, void* d_ws, size_t ws_size,
                              hipStream_t stream) {
  const float* x  = (const float*)d_in[0];
  const int*   ei = (const int*)d_in[1];
  const float* Wl_f1 = (const float*)d_in[2];
  const float* bl_f1 = (const float*)d_in[3];
  const float* Wr_f1 = (const float*)d_in[4];
  const float* Wl_b1 = (const float*)d_in[5];
  const float* bl_b1 = (const float*)d_in[6];
  const float* Wr_b1 = (const float*)d_in[7];
  const float* Wl_f2 = (const float*)d_in[8];
  const float* bl_f2 = (const float*)d_in[9];
  const float* Wr_f2 = (const float*)d_in[10];
  const float* Wl_b2 = (const float*)d_in[11];
  const float* bl_b2 = (const float*)d_in[12];
  const float* Wr_b2 = (const float*)d_in[13];
  const float* g1    = (const float*)d_in[14];
  const float* beta1 = (const float*)d_in[15];
  const float* g2    = (const float*)d_in[16];
  const float* beta2 = (const float*)d_in[17];

  const int nN = in_sizes[0] / 128;   // 50000 (divisible by 16)
  const int nE = in_sizes[1] / 2;     // 400000
  const int* src = ei;
  const int* dst = ei + nE;

  // workspace layout (floats; all sub-offsets keep 32B alignment)
  float* ws = (float*)d_ws;
  const size_t np = (size_t)nN * HID;
  float* aggF   = ws;                 // [nN,256] (layer1 uses [nN,128])
  float* aggB   = ws + np;
  float* H1     = ws + 2 * np;
  float* H2     = ws + 3 * np;
  float* cntF   = ws + 4 * np;        // [nN]
  float* cntB   = cntF + nN;          // [nN]
  float* gsum   = cntB + nN;          // [256]
  float* gsumsq = gsum + HID;         // [256]
  float* sA     = gsumsq + HID;       // [256]
  float* sBs    = sA + HID;           // [256]
  __bf16* pWlF1 = (__bf16*)(sBs + HID);        // 128*256 bf16 each
  __bf16* pWlB1 = pWlF1 + 128 * HID;
  __bf16* pWr1  = pWlB1 + 128 * HID;
  __bf16* pWlF2 = pWr1 + 128 * HID;            // 256*256 bf16 each
  __bf16* pWlB2 = pWlF2 + 256 * HID;
  __bf16* pWr2  = pWlB2 + 256 * HID;

  const int mTiles = nN / 16;
  const int statBlks = (nN + ROWS_PER_BLK - 1) / ROWS_PER_BLK;

  // ---- pack weights to bf16 WMMA fragment order (once; all independent) ----
  pack_b<128><<<(128 * HID + 255) / 256, 256, 0, stream>>>(Wl_f1, nullptr, pWlF1);
  pack_b<128><<<(128 * HID + 255) / 256, 256, 0, stream>>>(Wl_b1, nullptr, pWlB1);
  pack_b<128><<<(128 * HID + 255) / 256, 256, 0, stream>>>(Wr_f1, Wr_b1, pWr1);
  pack_b<256><<<(256 * HID + 255) / 256, 256, 0, stream>>>(Wl_f2, nullptr, pWlF2);
  pack_b<256><<<(256 * HID + 255) / 256, 256, 0, stream>>>(Wl_b2, nullptr, pWlB2);
  pack_b<256><<<(256 * HID + 255) / 256, 256, 0, stream>>>(Wr_f2, Wr_b2, pWr2);

  // ---- layer 1 ----
  hipMemsetAsync(cntF, 0, sizeof(float) * ((size_t)2 * nN + 4 * HID), stream);
  hipMemsetAsync(aggF, 0, sizeof(float) * (size_t)nN * 128, stream);
  hipMemsetAsync(aggB, 0, sizeof(float) * (size_t)nN * 128, stream);
  edge_count<<<(nE + 255) / 256, 256, 0, stream>>>(src, dst, cntF, cntB, nE);
  edge_scatter<128><<<nE, 128, 0, stream>>>(src, dst, x, aggF, aggB, nE);
  sage_gemm_wmma<128><<<mTiles, 256, 0, stream>>>(
      aggF, aggB, x, cntF, cntB, pWlF1, pWlB1, pWr1, bl_f1, bl_b1, H1);
  col_stats<<<statBlks, 256, 0, stream>>>(H1, gsum, gsumsq, nN);
  bn_finalize<<<1, 256, 0, stream>>>(gsum, gsumsq, g1, beta1, sA, sBs, nN);
  bn_apply_relu<<<2048, 256, 0, stream>>>(H1, sA, sBs, np);

  // ---- layer 2 ----
  hipMemsetAsync(aggF, 0, sizeof(float) * np, stream);
  hipMemsetAsync(aggB, 0, sizeof(float) * np, stream);
  hipMemsetAsync(gsum, 0, sizeof(float) * 2 * HID, stream);
  edge_scatter<256><<<nE, 256, 0, stream>>>(src, dst, H1, aggF, aggB, nE);
  sage_gemm_wmma<256><<<mTiles, 256, 0, stream>>>(
      aggF, aggB, H1, cntF, cntB, pWlF2, pWlB2, pWr2, bl_f2, bl_b2, H2);
  col_stats<<<statBlks, 256, 0, stream>>>(H2, gsum, gsumsq, nN);
  bn_finalize<<<1, 256, 0, stream>>>(gsum, gsumsq, g2, beta2, sA, sBs, nN);

  // ---- fused BN+ReLU+max epilogue ----
  hipMemsetAsync(d_out, 0, sizeof(float) * HID, stream);
  bn_relu_colmax<<<statBlks, 256, 0, stream>>>(H2, sA, sBs, (float*)d_out, nN);
}
